// ColNet2D_72962904425064
// MI455X (gfx1250) — matrix operand order repaired and checked by reading, not compiled, not run on hardware
//
#include <hip/hip_runtime.h>

typedef __attribute__((ext_vector_type(16))) _Float16 v16h;
typedef __attribute__((ext_vector_type(8)))  _Float16 v8h;
typedef __attribute__((ext_vector_type(8)))  float    v8f;

namespace {
constexpr int kB = 2;      // batch
constexpr int kF = 64;     // feature dim (= WMMA K total)
constexpr int kN = 4096;   // H*W
// temperature 1/0.07 folded together with log2(e): softmax done in base-2 so
// v_exp_f32 (natively 2^x) needs no pre-multiply in the hot loop.
constexpr float kScale = (1.0f / 0.07f) * 1.44269504088896340736f;
}

// ---------------------------------------------------------------------------
// RGB -> normalized LAB helper (one pixel, 3 channels)
// ---------------------------------------------------------------------------
__device__ __forceinline__ void rgb2lab3(float r, float g, float b,
                                         float& o0, float& o1, float& o2)
{
  float c[3] = {r, g, b};
#pragma unroll
  for (int j = 0; j < 3; ++j) {
    float v = c[j] + 0.5f;
    c[j] = (v > 0.04045f) ? __powf((v + 0.055f) * (1.0f / 1.055f), 2.4f)
                          : v * (1.0f / 12.92f);
  }
  float X = (0.412453f * c[0] + 0.357580f * c[1] + 0.180423f * c[2]) * (1.0f / 0.95047f);
  float Y =  0.212671f * c[0] + 0.715160f * c[1] + 0.072169f * c[2];
  float Z = (0.019334f * c[0] + 0.119193f * c[1] + 0.950227f * c[2]) * (1.0f / 1.08883f);
  float fx = (X > 0.008856f) ? cbrtf(X) : 7.787f * X + 16.0f / 116.0f;
  float fy = (Y > 0.008856f) ? cbrtf(Y) : 7.787f * Y + 16.0f / 116.0f;
  float fz = (Z > 0.008856f) ? cbrtf(Z) : 7.787f * Z + 16.0f / 116.0f;
  float L  = 116.0f * fy - 16.0f;
  float a  = 500.0f * (fx - fy);
  float bb = 200.0f * (fy - fz);
  o0 = (L - 50.0f) * (1.0f / 100.0f);
  o1 = a  * (1.0f / 110.0f);
  o2 = bb * (1.0f / 110.0f);
}

// Kernel 1: lab0 planar (B,3,N) for the loss; lab1 packed (B,N,float4) for
// a single wide load per attention column tile.
__global__ __launch_bounds__(256) void lab_kernel(
    const float* __restrict__ rgb0, const float* __restrict__ rgb1,
    float* __restrict__ lab0, float4* __restrict__ lab1p)
{
  int idx = blockIdx.x * blockDim.x + threadIdx.x;
  if (idx >= kB * kN) return;
  int b = idx / kN, n = idx % kN;
  size_t base = (size_t)b * 3 * kN + n;

  float o0, o1, o2;
  rgb2lab3(rgb0[base], rgb0[base + kN], rgb0[base + 2 * kN], o0, o1, o2);
  lab0[base]          = o0;
  lab0[base + kN]     = o1;
  lab0[base + 2 * kN] = o2;

  rgb2lab3(rgb1[base], rgb1[base + kN], rgb1[base + 2 * kN], o0, o1, o2);
  lab1p[(size_t)b * kN + n] = make_float4(o0, o1, o2, 0.0f);
}

// ---------------------------------------------------------------------------
// Kernel 2: (B,F,N) f32 -> (B,N,F) f16. Temperature * log2(e) folded into f0.
// ---------------------------------------------------------------------------
__global__ __launch_bounds__(256) void cvt_kernel(
    const float* __restrict__ f0, const float* __restrict__ f1,
    _Float16* __restrict__ f0h, _Float16* __restrict__ f1h)
{
  int idx = blockIdx.x * blockDim.x + threadIdx.x;
  if (idx >= kB * kF * kN) return;
  int b = idx / (kF * kN);
  int r = idx % (kF * kN);
  int f = r / kN, n = r % kN;
  size_t dsti = ((size_t)b * kN + n) * kF + f;
  f0h[dsti] = (_Float16)(f0[idx] * kScale);
  f1h[dsti] = (_Float16)f1[idx];
}

// ---------------------------------------------------------------------------
// Kernel 3: fused flash-attention + loss, one wave per 16-row query tile.
// Per column tile: scores(16x16,f32) via 2x v_wmma_f32_16x16x32_f16 (K=64).
// Softmax state is ENTIRELY per-lane (split-softmax, base-2): no cross-lane
// ops in the hot loop; one compare drives all selects. Unroll-4 provides four
// independent WMMA+softmax chains so the scheduler can fill the f16-WMMA
// RAW-hazard slots with real work instead of v_nops.
// ---------------------------------------------------------------------------
__global__ __launch_bounds__(128) void attn_kernel(
    const _Float16* __restrict__ f0h, const _Float16* __restrict__ f1h,
    const float* __restrict__ lab0, const float4* __restrict__ lab1p,
    float* __restrict__ out)
{
  const int wave = threadIdx.x >> 5;
  const int lane = threadIdx.x & 31;
  const int tile = blockIdx.x * 4 + wave;       // 0..511
  const int b    = tile >> 8;                   // tile / 256
  const int n0   = (tile & 255) << 4;           // query tile base row
  const int hg   = lane >> 4;                   // half-wave group (0/1)
  const int sub  = lane & 15;                   // row (A) or col (B/C) in tile

  // ---- A fragments: rows n0+sub, K = 0..63 ----
  // A k order chosen here: elem e <-> K = 16g+e (matches B)
  const _Float16* ap = f0h + ((size_t)b * kN + n0 + sub) * kF;
  v8h a00 = *(const v8h*)(ap +      16 * hg);
  v8h a01 = *(const v8h*)(ap +  8 + 16 * hg);
  v8h a10 = *(const v8h*)(ap + 32 + 16 * hg);
  v8h a11 = *(const v8h*)(ap + 40 + 16 * hg);
  v16h a0, a1;
#pragma unroll
  for (int i = 0; i < 8; ++i) {
    a0[i] = a00[i]; a0[i + 8] = a01[i];
    a1[i] = a10[i]; a1[i + 8] = a11[i];
  }

  // per-lane softmax state (8 rows each: r + 8*hg), base-2 log domain max
  float Mr[8], Sr[8], A0[8], A1[8], A2[8];
#pragma unroll
  for (int r = 0; r < 8; ++r) { Mr[r] = -1e30f; Sr[r] = 0.f; A0[r] = A1[r] = A2[r] = 0.f; }

  // B k order chosen here: elem e <-> K = 16g+e; this lane owns column sub.
  const _Float16* bp = f1h + ((size_t)b * kN + sub) * kF + 16 * hg;
  const float4*   lp = lab1p + (size_t)b * kN + sub;

#pragma unroll 4
  for (int m0 = 0; m0 < kN; m0 += 16) {
    v16h b0 = *(const v16h*)(bp);
    v16h b1 = *(const v16h*)(bp + 32);
    float4 lm = *lp;
    // prefetch one full unrolled iteration (4 tiles) ahead into near cache
    __builtin_prefetch(bp + 64 * kF, 0, 3);  // speculative; safe past the end
    bp += 16 * kF;
    lp += 16;

    v8f s = {};
    s = __builtin_amdgcn_wmma_f32_16x16x32_f16(false, a0, false, b0, (short)0, s, false, false);
    s = __builtin_amdgcn_wmma_f32_16x16x32_f16(false, a1, false, b1, (short)0, s, false, false);

#pragma unroll
    for (int r = 0; r < 8; ++r) {
      float v  = s[r];                 // logit in base-2 units (scale folded in A)
      float M  = Mr[r];
      bool  gt = v > M;                // one compare drives all four selects
      float nM = gt ? v : M;
      float mn = gt ? M : v;
      float ex = __builtin_amdgcn_exp2f(mn - nM); // exp2(-|v-M|); only factor != 1
      float al = gt ? ex : 1.0f;
      float p  = gt ? 1.0f : ex;
      Mr[r] = nM;
      Sr[r] = Sr[r] * al + p;
      A0[r] = A0[r] * al + p * lm.x;
      A1[r] = A1[r] * al + p * lm.y;
      A2[r] = A2[r] * al + p * lm.z;
    }
  }

  // ---- epilogue: split-softmax combine across the 16-lane group, loss ----
  const float* l0b = lab0 + (size_t)b * 3 * kN;
  float lossAcc = 0.f;
  const bool doOut = (sub == 0);
#pragma unroll
  for (int r = 0; r < 8; ++r) {
    // row max across lanes (masks 1,2,4,8 stay inside the 16-lane group)
    float Mg = Mr[r];
#pragma unroll
    for (int mask = 1; mask <= 8; mask <<= 1)
      Mg = fmaxf(Mg, __shfl_xor(Mg, mask));
    float sc = __builtin_amdgcn_exp2f(Mr[r] - Mg);
    float S  = Sr[r] * sc;
    float x0 = A0[r] * sc, x1 = A1[r] * sc, x2 = A2[r] * sc;
#pragma unroll
    for (int mask = 1; mask <= 8; mask <<= 1) {
      S  += __shfl_xor(S,  mask);
      x0 += __shfl_xor(x0, mask);
      x1 += __shfl_xor(x1, mask);
      x2 += __shfl_xor(x2, mask);
    }
    if (doOut) {
      int n = n0 + r + 8 * hg;
      float inv = 1.0f / S;
      float d0 = x0 * inv - l0b[0 * kN + n];
      float d1 = x1 * inv - l0b[1 * kN + n];
      float d2 = x2 * inv - l0b[2 * kN + n];
      float ad0 = fabsf(d0), ad1 = fabsf(d1), ad2 = fabsf(d2);
      float l1 = ad0 + ad1 + ad2;
      float hu = (ad0 < 1.f ? 0.5f * ad0 * ad0 : ad0 - 0.5f)
               + (ad1 < 1.f ? 0.5f * ad1 * ad1 : ad1 - 0.5f)
               + (ad2 < 1.f ? 0.5f * ad2 * ad2 : ad2 - 0.5f);
      lossAcc += l1 + hu;
    }
  }
  if (doOut) atomicAdd(out, lossAcc * (1.0f / (float)(kB * kN)));
}

// ---------------------------------------------------------------------------
// Launch
// ---------------------------------------------------------------------------
extern "C" void kernel_launch(void* const* d_in, const int* in_sizes, int n_in,
                              void* d_out, int out_size, void* d_ws, size_t ws_size,
                              hipStream_t stream)
{
  const float* rgb0 = (const float*)d_in[0];
  const float* rgb1 = (const float*)d_in[1];
  const float* f0   = (const float*)d_in[2];
  const float* f1   = (const float*)d_in[3];
  float* out        = (float*)d_out;

  // workspace: f0h (1 MB) | f1h (1 MB) | lab0 planar (96 KB) | lab1 packed (128 KB)
  _Float16* f0h = (_Float16*)d_ws;
  _Float16* f1h = f0h + (size_t)kB * kN * kF;
  float*  lab0  = (float*)(f1h + (size_t)kB * kN * kF);
  float4* lab1p = (float4*)(lab0 + (size_t)kB * 3 * kN);

  (void)hipMemsetAsync(d_out, 0, sizeof(float), stream);

  lab_kernel<<<(kB * kN + 255) / 256, 256, 0, stream>>>(rgb0, rgb1, lab0, lab1p);
  cvt_kernel<<<(kB * kF * kN + 255) / 256, 256, 0, stream>>>(f0, f1, f0h, f1h);
  // 512 waves total, 4 waves (128 threads) per block -> 128 blocks, exact.
  attn_kernel<<<128, 128, 0, stream>>>(f0h, f1h, lab0, lab1p, out);
}
